// VAEncoder_63041529971341
// MI455X (gfx1250) — matrix operand order repaired and checked by reading, not compiled, not run on hardware
//
#include <hip/hip_runtime.h>
#include <hip/hip_bf16.h>
#include <math.h>

// ---------------------------------------------------------------------------
// CDNA5 WMMA (wave32): A/B = 16x32 / 32x16 bf16 (8 VGPRs), C/D = 16x16 f32.
// ---------------------------------------------------------------------------
typedef __bf16 bf16_t;
typedef __attribute__((ext_vector_type(16))) __bf16 v16bf;
typedef __attribute__((ext_vector_type(8)))  __bf16 v8bf;
typedef __attribute__((ext_vector_type(8)))  float  v8f;

__device__ __forceinline__ bf16_t f2bf(float f) {
  union { float f; unsigned u; } v; v.f = f;
  unsigned r = v.u + 0x7FFFu + ((v.u >> 16) & 1u);   // round-to-nearest-even
  unsigned short h = (unsigned short)(r >> 16);
  return __builtin_bit_cast(__bf16, h);
}

__device__ __forceinline__ int refl(int i, int n) {
  if (i < 0) i = -i;
  if (i >= n) i = 2 * n - 2 - i;
  return i;
}

__device__ __forceinline__ v16bf cat16(v8bf lo, v8bf hi) {
  return __builtin_shufflevector(lo, hi, 0,1,2,3,4,5,6,7,8,9,10,11,12,13,14,15);
}

// ---------------------------------------------------------------------------
// Weight transform: fp32 (Cout,Cin,Kk,Kk) [mode 0] or (Cin,Cout,Kk,Kk) with
// spatial flip [mode 1, ConvTranspose] -> bf16 [ky*Kk+kx][CoutP][CinP],
// zero-padded so the conv kernel needs no bounds checks on A.
// ---------------------------------------------------------------------------
__global__ void k_wxform(const float* __restrict__ src, bf16_t* __restrict__ dst,
                         int Cout, int Cin, int CoutP, int CinP, int Kk, int mode,
                         int total)
{
  int i = blockIdx.x * blockDim.x + threadIdx.x;
  if (i >= total) return;
  int ci  = i % CinP;
  int t   = i / CinP;
  int co  = t % CoutP;
  int kyx = t / CoutP;
  int ky = kyx / Kk, kx = kyx - ky * Kk;
  float v = 0.f;
  if (co < Cout && ci < Cin) {
    long si;
    if (mode == 0) si = (((long)co * Cin + ci) * Kk + ky) * Kk + kx;
    else           si = (((long)ci * Cout + co) * Kk + (Kk - 1 - ky)) * Kk + (Kk - 1 - kx);
    v = src[si];
  }
  dst[i] = f2bf(v);
}

// Input: fp32 NCHW -> bf16 NHWC, channels zero-padded to CP
__global__ void k_in2nhwc(const float* __restrict__ src, bf16_t* __restrict__ dst,
                          int Bn, int C, int CP, int HW, int total)
{
  int i = blockIdx.x * blockDim.x + threadIdx.x;
  if (i >= total) return;
  int c   = i % CP;
  int pix = (i / CP) % HW;
  int b   = i / (CP * HW);
  float v = (c < C) ? src[((long)b * C + c) * HW + pix] : 0.f;
  dst[i] = f2bf(v);
}

__global__ void k_zero(float* __restrict__ p, int n) {
  int i = blockIdx.x * blockDim.x + threadIdx.x;
  if (i < n) p[i] = 0.0f;
}

// ---------------------------------------------------------------------------
// Implicit-GEMM conv via V_WMMA_F32_16X16X32_BF16, NHWC activations,
// 2x2 register blocking: each wave owns a 32(cout) x 32(pixel) tile
// (4 fp32 accumulators). Per (ky,kx,cin32) step: 2 A-frags + 2 B-frags
// (aligned b128 loads) -> 4 WMMAs.
// mode 1 = ConvTranspose2d(k3,s2,p1,op1) as gather over lhs-dilated input.
// ---------------------------------------------------------------------------
__global__ __launch_bounds__(128) void k_conv_wmma(
    const bf16_t* __restrict__ X, const bf16_t* __restrict__ Wg,
    const float* __restrict__ bias, float* __restrict__ Y,
    int Bn, int CinP, int H, int W, int Cout, int CoutP, int Kk,
    int stride, int pad, int Ho, int Wo, int mode, int pad_mode, int act)
{
  const int lane = threadIdx.x & 31;
  const int wave = threadIdx.x >> 5;
  const int coutBase = blockIdx.y << 5;              // 32 couts per tile row
  const int Npix = Bn * Ho * Wo;                     // multiple of 128 here
  const int pixBase = (blockIdx.x * 4 + wave) << 5;  // 32 pixels per wave
  if (pixBase >= Npix) return;                       // wave-uniform
  const bool has2 = (coutBase + 16) < CoutP;         // second cout tile valid?

  const int col  = lane & 15;
  const int HoWo = Ho * Wo;

  int b[2], oy[2], ox[2];
  #pragma unroll
  for (int t = 0; t < 2; ++t) {
    int p  = pixBase + t * 16 + col;
    b[t]   = p / HoWo;
    int rp = p - b[t] * HoWo;
    oy[t]  = rp / Wo;
    ox[t]  = rp - oy[t] * Wo;
  }

  const int arow  = lane & 15;                 // A row (cout within 16-tile)
  const int abase = (lane < 16) ? 0 : 8;       // A-K / D-M half selector
  const int bbase = (lane < 16) ? 0 : 8;       // B-K half selector
  const int cm0   = coutBase + arow;           // < CoutP (padded weights)

  v8f acc00 = {}, acc01 = {}, acc10 = {}, acc11 = {};

  for (int ky = 0; ky < Kk; ++ky) {
    int iy[2]; bool vy[2];
    #pragma unroll
    for (int t = 0; t < 2; ++t) {
      if (mode == 1) {
        int yd = oy[t] - pad + ky;
        vy[t] = (yd >= 0) && (yd <= 2 * H - 2) && ((yd & 1) == 0);
        iy[t] = yd >> 1;
      } else {
        int y = oy[t] * stride - pad + ky;
        if (pad_mode == 1) { y = refl(y, H); vy[t] = true; }
        else vy[t] = (y >= 0) && (y < H);
        iy[t] = y;
      }
    }
    for (int kx = 0; kx < Kk; ++kx) {
      bool v[2]; const bf16_t* Xrow[2];
      #pragma unroll
      for (int t = 0; t < 2; ++t) {
        int ix; bool vx;
        if (mode == 1) {
          int xd = ox[t] - pad + kx;
          vx = (xd >= 0) && (xd <= 2 * W - 2) && ((xd & 1) == 0);
          ix = xd >> 1;
        } else {
          ix = ox[t] * stride - pad + kx;
          if (pad_mode == 1) { ix = refl(ix, W); vx = true; }
          else vx = (ix >= 0) && (ix < W);
        }
        v[t] = vy[t] && vx;
        Xrow[t] = X + (((long)b[t] * H + (v[t] ? iy[t] : 0)) * W
                       + (v[t] ? ix : 0)) * CinP;
      }
      const bf16_t* __restrict__ W0 =
          Wg + ((long)(ky * Kk + kx) * CoutP + cm0) * CinP;
      const bf16_t* __restrict__ W1 = W0 + (long)16 * CinP;

      for (int cb = 0; cb < CinP; cb += 32) {
        // A fragments (padded weights: no guards)
        v16bf A0 = cat16(*(const v8bf*)(W0 + cb + abase),
                         *(const v8bf*)(W0 + cb + 16 + abase));
        // B fragments (zero when tap is in padding / dilation gap)
        v8bf z = {};
        v8bf b00 = v[0] ? *(const v8bf*)(Xrow[0] + cb + bbase)     : z;
        v8bf b01 = v[0] ? *(const v8bf*)(Xrow[0] + cb + bbase + 16) : z;
        v8bf b10 = v[1] ? *(const v8bf*)(Xrow[1] + cb + bbase)     : z;
        v8bf b11 = v[1] ? *(const v8bf*)(Xrow[1] + cb + bbase + 16) : z;
        v16bf B0 = cat16(b00, b01);
        v16bf B1 = cat16(b10, b11);

        acc00 = __builtin_amdgcn_wmma_f32_16x16x32_bf16(false, A0, false, B0,
                                                        (short)0, acc00, false, false);
        acc01 = __builtin_amdgcn_wmma_f32_16x16x32_bf16(false, A0, false, B1,
                                                        (short)0, acc01, false, false);
        if (has2) {                                  // wave-uniform branch
          v16bf A1 = cat16(*(const v8bf*)(W1 + cb + abase),
                           *(const v8bf*)(W1 + cb + 16 + abase));
          acc10 = __builtin_amdgcn_wmma_f32_16x16x32_bf16(false, A1, false, B0,
                                                          (short)0, acc10, false, false);
          acc11 = __builtin_amdgcn_wmma_f32_16x16x32_bf16(false, A1, false, B1,
                                                          (short)0, acc11, false, false);
        }
      }
      // speculative prefetch of the next tap's weight row (global_prefetch_b8)
      __builtin_prefetch(W0 + (long)CoutP * CinP, 0, 0);
    }
  }

  // Epilogue: contiguous output channels per lane in NHWC fp32
  #pragma unroll
  for (int t = 0; t < 2; ++t) {
    float* __restrict__ Yp = Y + (((long)b[t] * Ho + oy[t]) * Wo + ox[t]) * Cout;
    const v8f a0 = t ? acc01 : acc00;
    #pragma unroll
    for (int rr = 0; rr < 8; ++rr) {
      const int cm = coutBase + abase + rr;      // D: VGPR rr -> M = rr + half*8
      if (cm < Cout) {
        float y = a0[rr] + bias[cm];
        if (act == 1) y = tanhf(y);
        Yp[cm] = y;
      }
    }
    if (has2) {
      const v8f a1 = t ? acc11 : acc10;
      #pragma unroll
      for (int rr = 0; rr < 8; ++rr) {
        const int cm = coutBase + 16 + abase + rr;
        if (cm < Cout) {
          float y = a1[rr] + bias[cm];
          if (act == 1) y = tanhf(y);
          Yp[cm] = y;
        }
      }
    }
  }
}

// ---------------------------------------------------------------------------
// BatchNorm (training-mode batch stats) over NHWC: one block per channel.
// stats[2c] = mean, stats[2c+1] = rsqrt(var + eps)
// ---------------------------------------------------------------------------
__global__ __launch_bounds__(256) void k_bn_stats(
    const float* __restrict__ X, float* __restrict__ stats, int C, long NP)
{
  const int c = blockIdx.x;
  const int t = threadIdx.x;
  float s = 0.f, s2 = 0.f;
  for (long n = t; n < NP; n += blockDim.x) {
    float v = X[n * C + c]; s += v; s2 += v * v;
  }
  __shared__ float sh[256], sh2[256];
  sh[t] = s; sh2[t] = s2;
  __syncthreads();
  for (int off = 128; off > 0; off >>= 1) {
    if (t < off) { sh[t] += sh[t + off]; sh2[t] += sh2[t + off]; }
    __syncthreads();
  }
  if (t == 0) {
    float n = (float)NP;
    float mean = sh[0] / n;
    float var = sh2[0] / n - mean * mean;
    stats[2 * c]     = mean;
    stats[2 * c + 1] = rsqrtf(var + 1e-5f);
  }
}

// normalize + ReLU + bf16 cast, NHWC -> NHWC (next conv input)
__global__ void k_bn_relu(const float* __restrict__ X, const float* __restrict__ stats,
                          bf16_t* __restrict__ Y, int C, long total)
{
  long i = (long)blockIdx.x * blockDim.x + threadIdx.x;
  if (i >= total) return;
  int c = (int)(i % C);
  float v = (X[i] - stats[2 * c]) * stats[2 * c + 1];
  v = v > 0.f ? v : 0.f;
  Y[i] = f2bf(v);
}

// ---------------------------------------------------------------------------
// Segment mean: accumulate with GLOBAL_ATOMIC_ADD_F32, then scatter means.
// MU is NHWC fp32 (C=8 contiguous per pixel); output is NCHW fp32.
// ---------------------------------------------------------------------------
__global__ void k_seg_accum(const float* __restrict__ MU, const int* __restrict__ inst,
                            float* __restrict__ sums, float* __restrict__ cnts,
                            int Bn, int C, int HW, int KI)
{
  int i = blockIdx.x * blockDim.x + threadIdx.x;   // over Bn*HW pixels
  if (i >= Bn * HW) return;
  int bb  = i / HW;
  int seg = bb * KI + inst[i];
  const float* mp = MU + (long)i * C;
  for (int c = 0; c < C; ++c) atomicAdd(&sums[seg * C + c], mp[c]);
  atomicAdd(&cnts[seg], 1.0f);
}

__global__ void k_seg_scatter(const float* __restrict__ sums, const float* __restrict__ cnts,
                              const int* __restrict__ inst, float* __restrict__ out,
                              int Bn, int C, int HW, int KI)
{
  long i = (long)blockIdx.x * blockDim.x + threadIdx.x; // over Bn*C*HW (NCHW)
  if (i >= (long)Bn * C * HW) return;
  int bb  = (int)(i / ((long)C * HW));
  long r  = i - (long)bb * C * HW;
  int c   = (int)(r / HW);
  int pix = (int)(r - (long)c * HW);
  int seg = bb * KI + inst[bb * HW + pix];
  float cnt = cnts[seg]; if (cnt < 1.f) cnt = 1.f;
  out[i] = sums[seg * C + c] / cnt;
}

// ---------------------------------------------------------------------------
// Host orchestration
// ---------------------------------------------------------------------------
extern "C" void kernel_launch(void* const* d_in, const int* in_sizes, int n_in,
                              void* d_out, int out_size, void* d_ws, size_t ws_size,
                              hipStream_t stream)
{
  (void)in_sizes; (void)n_in; (void)out_size; (void)ws_size;
  const int Bn = 4, KI = 20, OUTC = 8;

  const float* x    = (const float*)d_in[0];
  const int*   inst = (const int*)d_in[1];
  const float* w_in = (const float*)d_in[2];
  const float* b_in = (const float*)d_in[3];
  const float *dw[4], *db[4], *uw[4], *ub[4];
  for (int i = 0; i < 4; ++i) { dw[i] = (const float*)d_in[4 + 2 * i];  db[i] = (const float*)d_in[5 + 2 * i]; }
  for (int i = 0; i < 4; ++i) { uw[i] = (const float*)d_in[12 + 2 * i]; ub[i] = (const float*)d_in[13 + 2 * i]; }
  const float* w_fc = (const float*)d_in[20];
  const float* b_fc = (const float*)d_in[21];

  // ---- workspace carve (256B aligned) ----
  char* ws = (char*)d_ws;
  size_t off = 0;
  auto carve = [&](size_t bytes) -> void* {
    void* pp = ws + off;
    off = (off + bytes + 255) & ~(size_t)255;
    return pp;
  };
  // padded bf16 weight buffers: [KK][CoutP][CinP]
  const int nw_in = 49 * 32 * 32;                                   // CinP 3->32
  const int ndw[4] = { 9 * 64 * 32, 9 * 128 * 64, 9 * 256 * 128, 9 * 512 * 256 };
  const int nuw[4] = { 9 * 256 * 512, 9 * 128 * 256, 9 * 64 * 128, 9 * 32 * 64 };
  const int nw_fc = 49 * 16 * 32;                                   // CoutP 8->16

  bf16_t* bw_in = (bf16_t*)carve(sizeof(bf16_t) * nw_in);
  bf16_t* bdw[4]; for (int i = 0; i < 4; ++i) bdw[i] = (bf16_t*)carve(sizeof(bf16_t) * ndw[i]);
  bf16_t* buw[4]; for (int i = 0; i < 4; ++i) buw[i] = (bf16_t*)carve(sizeof(bf16_t) * nuw[i]);
  bf16_t* bw_fc = (bf16_t*)carve(sizeof(bf16_t) * nw_fc);

  const long maxAct = (long)Bn * 256 * 256 * 32;   // 8.4M elements (NHWC)
  bf16_t* actA   = (bf16_t*)carve(sizeof(bf16_t) * maxAct);
  float*  f32tmp = (float*)carve(sizeof(float) * maxAct);
  float*  stats  = (float*)carve(sizeof(float) * 1024);
  float*  segS   = (float*)carve(sizeof(float) * Bn * KI * OUTC);
  float*  segC   = (float*)carve(sizeof(float) * Bn * KI);

  auto wxf = [&](const float* src, bf16_t* dst, int Cout, int Cin, int CoutP, int CinP,
                 int Kk, int mode) {
    int total = Kk * Kk * CoutP * CinP;
    k_wxform<<<(total + 255) / 256, 256, 0, stream>>>(src, dst, Cout, Cin, CoutP, CinP,
                                                      Kk, mode, total);
  };
  auto conv = [&](const bf16_t* X, const bf16_t* Wg, const float* bias, float* Y,
                  int CinP, int H, int Cout, int CoutP, int Kk, int stride, int pad,
                  int Ho, int mode, int pad_mode, int act) {
    int Npix = Bn * Ho * Ho;
    dim3 grid(Npix / 128, (CoutP + 31) / 32);
    k_conv_wmma<<<grid, 128, 0, stream>>>(X, Wg, bias, Y, Bn, CinP, H, H, Cout, CoutP,
                                          Kk, stride, pad, Ho, Ho, mode, pad_mode, act);
  };
  auto bn = [&](const float* X, bf16_t* Y, int C, long NP) {
    k_bn_stats<<<C, 256, 0, stream>>>(X, stats, C, NP);
    long total = NP * C;
    k_bn_relu<<<(unsigned)((total + 255) / 256), 256, 0, stream>>>(X, stats, Y, C, total);
  };

  // ---- weight transforms + input layout conversion ----
  wxf(w_in, bw_in, 32, 3, 32, 32, 7, 0);
  {
    int C = 32;
    for (int i = 0; i < 4; ++i) { wxf(dw[i], bdw[i], 2 * C, C, 2 * C, C, 3, 0); C *= 2; }
    for (int i = 0; i < 4; ++i) { wxf(uw[i], buw[i], C / 2, C, C / 2, C, 3, 1); C /= 2; }
  }
  wxf(w_fc, bw_fc, OUTC, 32, 16, 32, 7, 0);
  {
    int total = Bn * 256 * 256 * 32;
    k_in2nhwc<<<(total + 255) / 256, 256, 0, stream>>>(x, actA, Bn, 3, 32, 256 * 256, total);
  }

  // ---- backbone ----
  conv(actA, bw_in, b_in, f32tmp, 32, 256, 32, 32, 7, 1, 3, 256, 0, /*reflect*/1, 0);
  bn(f32tmp, actA, 32, (long)Bn * 256 * 256);

  int C = 32, H = 256;
  for (int i = 0; i < 4; ++i) {                 // downsample
    int Co = C * 2, Ho = H / 2;
    conv(actA, bdw[i], db[i], f32tmp, C, H, Co, Co, 3, 2, 1, Ho, 0, 0, 0);
    bn(f32tmp, actA, Co, (long)Bn * Ho * Ho);
    C = Co; H = Ho;
  }
  for (int i = 0; i < 4; ++i) {                 // upsample (transposed conv)
    int Co = C / 2, Ho = H * 2;
    conv(actA, buw[i], ub[i], f32tmp, C, H, Co, Co, 3, 1, 1, Ho, /*mode*/1, 0, 0);
    bn(f32tmp, actA, Co, (long)Bn * Ho * Ho);
    C = Co; H = Ho;
  }
  conv(actA, bw_fc, b_fc, f32tmp, 32, 256, OUTC, 16, 7, 1, 3, 256, 0, /*reflect*/1, /*tanh*/1);

  // ---- segment mean + scatter ----
  int nSeg = Bn * KI;
  k_zero<<<(nSeg * OUTC + 255) / 256, 256, 0, stream>>>(segS, nSeg * OUTC);
  k_zero<<<(nSeg + 255) / 256, 256, 0, stream>>>(segC, nSeg);
  int HW = 256 * 256;
  k_seg_accum<<<(Bn * HW + 255) / 256, 256, 0, stream>>>(f32tmp, inst, segS, segC, Bn, OUTC, HW, KI);
  long totalOut = (long)Bn * OUTC * HW;
  k_seg_scatter<<<(unsigned)((totalOut + 255) / 256), 256, 0, stream>>>(segS, segC, inst,
                                                                       (float*)d_out, Bn, OUTC, HW, KI);
}
